// EdgeModel_39591008534980
// MI455X (gfx1250) — matrix operand order repaired and checked by reading, not compiled, not run on hardware
//
#include <hip/hip_runtime.h>
#include <stdint.h>

#define D_NODE   128
#define D_EDGE   128
#define D_IN     384          // 2*D_NODE + D_EDGE
#define D_HID    512
#define D_OUT    128

#define M_TILE   64           // edges per workgroup
#define ASTR     (D_IN + 8)   // 392 bf16: row stride -> 4-bank rotation
#define HSTR     (D_HID + 8)  // 520 bf16: row stride -> 4-bank rotation

typedef __attribute__((ext_vector_type(16))) __bf16 v16bf;
typedef __attribute__((ext_vector_type(8)))  float  v8f;

union Frag {
    v16bf v;
    uint4 q[2];
};

__device__ __forceinline__ unsigned short f2bf(float f) {
    unsigned u = __float_as_uint(f);
    u += 0x7FFFu + ((u >> 16) & 1u);      // round-to-nearest-even
    return (unsigned short)(u >> 16);
}

// W1 [384,512] row-major fp32  ->  W1t [512][384] bf16 (B-fragment friendly: column-major)
__global__ void prep_w1(const float* __restrict__ W1, unsigned short* __restrict__ W1t) {
    int i = blockIdx.x * blockDim.x + threadIdx.x;
    if (i >= D_HID * D_IN) return;
    int n = i / D_IN, k = i % D_IN;
    W1t[(size_t)n * D_IN + k] = f2bf(W1[(size_t)k * D_HID + n]);
}

// W2 [512,128] row-major fp32  ->  W2t [128][512] bf16
__global__ void prep_w2(const float* __restrict__ W2, unsigned short* __restrict__ W2t) {
    int i = blockIdx.x * blockDim.x + threadIdx.x;
    if (i >= D_OUT * D_HID) return;
    int n = i / D_HID, k = i % D_HID;
    W2t[(size_t)n * D_HID + k] = f2bf(W2[(size_t)k * D_OUT + n]);
}

__launch_bounds__(256, 1)
__global__ void edge_mlp_kernel(const float* __restrict__ x,
                                const int*   __restrict__ eidx,   // [2,E]
                                const float* __restrict__ eattr,  // [E,128]
                                const unsigned short* __restrict__ W1t,
                                const float* __restrict__ b1,
                                const unsigned short* __restrict__ W2t,
                                const float* __restrict__ b2,
                                float* __restrict__ out,          // [E,128]
                                int E)
{
    // single LDS buffer: holds A (64 x ASTR) for layer1, then overlaid H (64 x HSTR)
    __shared__ __align__(16) unsigned short lds[M_TILE * HSTR];   // 66560 B

    const int tid   = threadIdx.x;
    const int lane  = tid & 31;
    const int wv    = tid >> 5;          // wave 0..7
    const int hi    = lane >> 4;         // half-wave 0/1
    const int ln    = lane & 15;
    const int ebase = blockIdx.x * M_TILE;

    // ---------------- gather + fp32->bf16 into LDS ----------------
    // 64 edges * 96 float4 (384 floats) each; 24 float4 per thread
    for (int i = tid; i < M_TILE * 96; i += 256) {
        int e = i / 96;
        int c = i % 96;                  // float4 column within 384
        int eg = ebase + e;
        if (eg >= E) eg = 0;             // clamp (garbage rows masked at store)
        const float4* src;
        if (c < 32) {
            int r = eidx[eg];                                   // source node
            src = (const float4*)(x + (size_t)r * D_NODE) + c;
        } else if (c < 64) {
            int r = eidx[E + eg];                               // target node
            src = (const float4*)(x + (size_t)r * D_NODE) + (c - 32);
        } else {
            src = (const float4*)(eattr + (size_t)eg * D_EDGE) + (c - 64);
        }
        float4 f = *src;
        unsigned short* dst = &lds[e * ASTR + c * 4];
        dst[0] = f2bf(f.x); dst[1] = f2bf(f.y);
        dst[2] = f2bf(f.z); dst[3] = f2bf(f.w);
    }
    __syncthreads();

    // ---------------- layer 1: H = relu(A @ W1 + b1) ----------------
    // wave wv owns N slice [64*wv, 64*wv+64): 4 Mtiles x 4 Ntiles of 16x16
    v8f acc[4][4];
    {
        v8f z = {0.f,0.f,0.f,0.f,0.f,0.f,0.f,0.f};
#pragma unroll
        for (int mt = 0; mt < 4; ++mt)
#pragma unroll
            for (int nt = 0; nt < 4; ++nt) acc[mt][nt] = z;
    }
    const int nw = wv * 64;
    for (int kk = 0; kk < D_IN; kk += 32) {
        Frag a[4];
        const int akb = kk + hi * 8;     // A layout: lanes0-15 K{0-7,16-23}, lanes16-31 K{8-15,24-31}
#pragma unroll
        for (int mt = 0; mt < 4; ++mt) {
            const unsigned short* ap = &lds[(ln + 16 * mt) * ASTR + akb];
            a[mt].q[0] = *(const uint4*)(ap);
            a[mt].q[1] = *(const uint4*)(ap + 16);
        }
        const int bkb = kk + hi * 8;     // B layout: lanes0-15 K{0-7,16-23}, lanes16-31 K{8-15,24-31}
#pragma unroll
        for (int nt = 0; nt < 4; ++nt) {
            Frag b;
            const unsigned short* bp = &W1t[(size_t)(nw + 16 * nt + ln) * D_IN + bkb];
            b.q[0] = *(const uint4*)(bp);
            b.q[1] = *(const uint4*)(bp + 16);
#pragma unroll
            for (int mt = 0; mt < 4; ++mt)
                acc[mt][nt] = __builtin_amdgcn_wmma_f32_16x16x32_bf16(
                    false, a[mt].v, false, b.v, (short)0, acc[mt][nt], false, false);
        }
    }
    __syncthreads();                     // all A reads done before H overlays the buffer

    // bias + relu + store H (bf16) into LDS
#pragma unroll
    for (int nt = 0; nt < 4; ++nt) {
        const int n = nw + 16 * nt + ln;
        const float bias = b1[n];
#pragma unroll
        for (int mt = 0; mt < 4; ++mt) {
#pragma unroll
            for (int v = 0; v < 8; ++v) {
                float h = acc[mt][nt][v] + bias;   // C layout: M = v + 8*hi, N = ln
                h = h > 0.f ? h : 0.f;
                lds[(16 * mt + v + 8 * hi) * HSTR + n] = f2bf(h);
            }
        }
    }
    __syncthreads();

    // ---------------- layer 2: O = H @ W2 + b2 ----------------
    // wave wv owns output N slice [16*wv, 16*wv+16): 4 Mtiles x 1 Ntile
    v8f oacc[4];
    {
        v8f z = {0.f,0.f,0.f,0.f,0.f,0.f,0.f,0.f};
#pragma unroll
        for (int mt = 0; mt < 4; ++mt) oacc[mt] = z;
    }
    const int n2 = wv * 16 + ln;         // 0..127
    for (int kk = 0; kk < D_HID; kk += 32) {
        Frag a[4];
        const int akb = kk + hi * 8;
#pragma unroll
        for (int mt = 0; mt < 4; ++mt) {
            const unsigned short* ap = &lds[(ln + 16 * mt) * HSTR + akb];
            a[mt].q[0] = *(const uint4*)(ap);
            a[mt].q[1] = *(const uint4*)(ap + 16);
        }
        Frag b;
        const unsigned short* bp = &W2t[(size_t)n2 * D_HID + kk + hi * 8];
        b.q[0] = *(const uint4*)(bp);
        b.q[1] = *(const uint4*)(bp + 16);
#pragma unroll
        for (int mt = 0; mt < 4; ++mt)
            oacc[mt] = __builtin_amdgcn_wmma_f32_16x16x32_bf16(
                false, a[mt].v, false, b.v, (short)0, oacc[mt], false, false);
    }

    const float bias2 = b2[n2];
#pragma unroll
    for (int mt = 0; mt < 4; ++mt) {
#pragma unroll
        for (int v = 0; v < 8; ++v) {
            const int m  = 16 * mt + v + 8 * hi;
            const int eg = ebase + m;
            if (eg < E)
                out[(size_t)eg * D_OUT + n2] = oacc[mt][v] + bias2;
        }
    }
}

extern "C" void kernel_launch(void* const* d_in, const int* in_sizes, int n_in,
                              void* d_out, int out_size, void* d_ws, size_t ws_size,
                              hipStream_t stream) {
    const float* x    = (const float*)d_in[0];
    const int*   eidx = (const int*)  d_in[1];
    const float* eatt = (const float*)d_in[2];
    const float* W1   = (const float*)d_in[3];
    const float* b1   = (const float*)d_in[4];
    const float* W2   = (const float*)d_in[5];
    const float* b2   = (const float*)d_in[6];
    float* out = (float*)d_out;

    const int E = in_sizes[1] / 2;       // edge_index is [2, E]

    unsigned short* W1t = (unsigned short*)d_ws;                     // 512*384 bf16 = 384 KB
    unsigned short* W2t = W1t + (size_t)D_HID * D_IN;                // 128*512 bf16 = 128 KB

    prep_w1<<<(D_HID * D_IN + 255) / 256, 256, 0, stream>>>(W1, W1t);
    prep_w2<<<(D_OUT * D_HID + 255) / 256, 256, 0, stream>>>(W2, W2t);

    const int blocks = (E + M_TILE - 1) / M_TILE;
    edge_mlp_kernel<<<blocks, 256, 0, stream>>>(x, eidx, eatt, W1t, b1, W2t, b2, out, E);
}
